// Transformer_51256139710912
// MI455X (gfx1250) — matrix operand order repaired and checked by reading, not compiled, not run on hardware
//
#include <hip/hip_runtime.h>
#include <cstdint>
#include <cstddef>

// ---------------------------------------------------------------------------
// CDNA5 wave32 WMMA types: D(f32 16x16) = A(bf16 16x32) x B(bf16 32x16) + C
// ---------------------------------------------------------------------------
typedef __bf16 bf16_t;
typedef bf16_t bf16x16 __attribute__((ext_vector_type(16)));
typedef bf16_t bf16x8  __attribute__((ext_vector_type(8)));
typedef bf16_t bf16x4  __attribute__((ext_vector_type(4)));
typedef float  f32x8   __attribute__((ext_vector_type(8)));
typedef int    i32x4   __attribute__((ext_vector_type(4)));

union frag16 { bf16x16 f; bf16x8 h[2]; };

__device__ __forceinline__ bf16_t f2bf(float f) {
  uint32_t u = __builtin_bit_cast(uint32_t, f);
  uint32_t r = (u + 0x7FFFu + ((u >> 16) & 1u)) >> 16;   // RNE
  return __builtin_bit_cast(bf16_t, (unsigned short)r);
}

// store helpers (fp32 or bf16 output)
__device__ __forceinline__ void st1(float* p, float v)  { *p = v; }
__device__ __forceinline__ void st1(bf16_t* p, float v) { *p = f2bf(v); }
__device__ __forceinline__ void st4(float* p, float4 v) { *(float4*)p = v; }
__device__ __forceinline__ void st4(bf16_t* p, float4 v) {
  bf16x4 o = { f2bf(v.x), f2bf(v.y), f2bf(v.z), f2bf(v.w) };
  *(bf16x4*)p = o;
}

// ---- problem constants ----
constexpr int Dm = 1024, Hh = 16, DH = 64, Bb = 4, Ss = 1024, Mr = Bb * Ss;
constexpr float EPS = 1e-5f;

// ---------------------------------------------------------------------------
// 16-lane xor reductions: v_permlane16_b32 (VALU) when available, else shfl.
// ---------------------------------------------------------------------------
__device__ __forceinline__ float lanex16(float v, uint32_t s0, uint32_t s1, int mask) {
#if defined(__gfx1250__) && __has_builtin(__builtin_amdgcn_permlane16)
  (void)mask;
  return __builtin_bit_cast(float, __builtin_amdgcn_permlane16(
      __builtin_bit_cast(unsigned, v), __builtin_bit_cast(unsigned, v), s0, s1, false, false));
#else
  (void)s0; (void)s1;
  return __shfl_xor(v, mask, 32);
#endif
}
__device__ __forceinline__ float redmax16(float v) {
  v = fmaxf(v, lanex16(v, 0x67452301u, 0xEFCDAB89u, 1));
  v = fmaxf(v, lanex16(v, 0x54761032u, 0xDCFE98BAu, 2));
  v = fmaxf(v, lanex16(v, 0x32107654u, 0xBA98FEDCu, 4));
  v = fmaxf(v, lanex16(v, 0xFEDCBA98u, 0x76543210u, 8));
  return v;
}
__device__ __forceinline__ float redsum16(float v) {
  v += lanex16(v, 0x67452301u, 0xEFCDAB89u, 1);
  v += lanex16(v, 0x54761032u, 0xDCFE98BAu, 2);
  v += lanex16(v, 0x32107654u, 0xBA98FEDCu, 4);
  v += lanex16(v, 0xFEDCBA98u, 0x76543210u, 8);
  return v;
}

// ---------------------------------------------------------------------------
// Async global->LDS (ASYNCcnt path). Builtin takes (int4 AS1*, int4 AS3*,
// imm offset, cpol) -- clang prints AS1 as `__device__`, AS3 as `__shared__`.
// ---------------------------------------------------------------------------
#if defined(__gfx1250__) && __has_builtin(__builtin_amdgcn_global_load_async_to_lds_b128)
#define HAVE_ASYNC 1
typedef __attribute__((address_space(1))) i32x4 gv4i_t;
typedef __attribute__((address_space(3))) i32x4 lv4i_t;
__device__ __forceinline__ void async_cp16(const void* g, void* l) {
  __builtin_amdgcn_global_load_async_to_lds_b128(
      (gv4i_t*)const_cast<void*>(g), (lv4i_t*)l, 0, 0);
}
__device__ __forceinline__ void async_wait0() {
#if __has_builtin(__builtin_amdgcn_s_wait_asynccnt)
  __builtin_amdgcn_s_wait_asynccnt(0);
#else
  asm volatile("s_wait_asynccnt 0x0" ::: "memory");
#endif
}
#else
#define HAVE_ASYNC 0
__device__ __forceinline__ void async_cp16(const void* g, void* l) {
  *(bf16x8*)l = *(const bf16x8*)g;
}
__device__ __forceinline__ void async_wait0() {}
#endif

// ===========================================================================
// fp32 -> bf16 convert (weights), 4 elems/thread
// ===========================================================================
__global__ __launch_bounds__(256)
void cvt_kernel(const float* __restrict__ in, bf16_t* __restrict__ out, int n) {
  int i = (blockIdx.x * 256 + threadIdx.x) * 4;
  if (i < n) {
    float4 f = *(const float4*)(in + i);
    st4(out + i, f);
  }
}

// ===========================================================================
// LayerNorm: one block (256 thr) per row; fp32 in, OutT out
// ===========================================================================
template <typename OutT>
__global__ __launch_bounds__(256)
void ln_kernel(const float* __restrict__ x, const float* __restrict__ g,
               const float* __restrict__ bt, OutT* __restrict__ out) {
  const int row = blockIdx.x, tid = threadIdx.x;
  const int lane = tid & 31, wave = tid >> 5;
  const float* xr = x + (size_t)row * Dm;
  const int c = tid * 4;

  float4 f = *(const float4*)(xr + c);
  float s  = f.x + f.y + f.z + f.w;
  float s2 = f.x * f.x + f.y * f.y + f.z * f.z + f.w * f.w;
  #pragma unroll
  for (int m = 1; m < 32; m <<= 1) { s += __shfl_xor(s, m, 32); s2 += __shfl_xor(s2, m, 32); }

  __shared__ float r1[8], r2[8];
  if (lane == 0) { r1[wave] = s; r2[wave] = s2; }
  __syncthreads();
  float t1 = 0.f, t2 = 0.f;
  #pragma unroll
  for (int i = 0; i < 8; ++i) { t1 += r1[i]; t2 += r2[i]; }
  const float mean = t1 * (1.0f / Dm);
  const float var  = t2 * (1.0f / Dm) - mean * mean;
  const float inv  = rsqrtf(var + EPS);

  float4 gg = *(const float4*)(g + c);
  float4 bb = *(const float4*)(bt + c);
  float4 o;
  o.x = (f.x - mean) * inv * gg.x + bb.x;
  o.y = (f.y - mean) * inv * gg.y + bb.y;
  o.z = (f.z - mean) * inv * gg.z + bb.z;
  o.w = (f.w - mean) * inv * gg.w + bb.w;
  st4(out + (size_t)row * Dm + c, o);
}

// ===========================================================================
// GEMM: out = act((Xb[M,K] @ Wb[N,K]^T + bias) * alpha) (+ res)
// bf16 inputs, tile 128x64, 8 waves x (32x32 = 2x2 WMMA)
// ===========================================================================
constexpr int BM = 128, BN = 64, BK = 32;

template <int ACT, bool RES, typename OutT>
__global__ __launch_bounds__(256)
void gemm_kernel(const bf16_t* __restrict__ Xb, const bf16_t* __restrict__ Wb,
                 const float* __restrict__ bias, const float* __restrict__ res,
                 OutT* __restrict__ out, int M, int N, int K, float alpha) {
  __shared__ bf16_t As[BM][BK + 8];      // row stride 80B (16B-aligned)
  __shared__ bf16_t Bs[BN][BK + 8];

  const int tid = threadIdx.x;
  const int lane = tid & 31, wave = tid >> 5;
  const int wm = wave & 3, wn = wave >> 2;
  const int half = lane >> 4, l16 = lane & 15;
  const int row0 = blockIdx.y * BM, col0 = blockIdx.x * BN;

  f32x8 acc[2][2] = {};

  for (int k0 = 0; k0 < K; k0 += BK) {
    __syncthreads();
    // A tile 128x32 bf16: 512 16B-chunks, 2/thread
    #pragma unroll
    for (int t = 0; t < 2; ++t) {
      int slot = tid + t * 256;
      int m = slot >> 2, c8 = (slot & 3) << 3;
      async_cp16(Xb + (size_t)(row0 + m) * K + k0 + c8, &As[m][c8]);
    }
    // B tile 64x32 bf16: 256 chunks, 1/thread
    {
      int n = tid >> 2, c8 = (tid & 3) << 3;
      async_cp16(Wb + (size_t)(col0 + n) * K + k0 + c8, &Bs[n][c8]);
    }
    async_wait0();
    __syncthreads();

    frag16 a[2], b[2];
    #pragma unroll
    for (int i = 0; i < 2; ++i) {
      int m = wm * 32 + i * 16 + l16;
      a[i].h[0] = *(const bf16x8*)&As[m][8 * half];
      a[i].h[1] = *(const bf16x8*)&As[m][16 + 8 * half];
    }
    #pragma unroll
    for (int j = 0; j < 2; ++j) {
      int n = wn * 32 + j * 16 + l16;
      b[j].h[0] = *(const bf16x8*)&Bs[n][8 * half];
      b[j].h[1] = *(const bf16x8*)&Bs[n][16 + 8 * half];
    }
    #pragma unroll
    for (int i = 0; i < 2; ++i)
      #pragma unroll
      for (int j = 0; j < 2; ++j)
        acc[i][j] = __builtin_amdgcn_wmma_f32_16x16x32_bf16(
            false, a[i].f, false, b[j].f, (short)0, acc[i][j], false, false);
  }

  #pragma unroll
  for (int i = 0; i < 2; ++i)
    #pragma unroll
    for (int j = 0; j < 2; ++j) {
      const int col = col0 + wn * 32 + j * 16 + l16;
      const float bv = bias[col];
      #pragma unroll
      for (int r = 0; r < 8; ++r) {
        const int row = row0 + wm * 32 + i * 16 + r + 8 * half;
        float v = (acc[i][j][r] + bv) * alpha;
        if (ACT == 1) v = v > 0.f ? v : 0.f;
        if (RES) v += res[(size_t)row * N + col];
        st1(out + (size_t)row * N + col, v);
      }
    }
}

// ===========================================================================
// Flash attention (no mask), bf16 Q/K/V (Q pre-scaled by 1/sqrt(DH)), bf16 out
// 1 workgroup = 128 q rows of one (b,h); K [kv][dh] in LDS, V transposed.
// ===========================================================================
__global__ __launch_bounds__(256)
void flash_attn_kernel(const bf16_t* __restrict__ Q, const bf16_t* __restrict__ K,
                       const bf16_t* __restrict__ V, bf16_t* __restrict__ O,
                       int Sq, int Skv) {
  const int bh = blockIdx.y;
  const int b = bh / Hh, h = bh % Hh;
  const size_t qbase = (size_t)b * Sq * Dm + (size_t)h * DH;
  const size_t kbase = (size_t)b * Skv * Dm + (size_t)h * DH;

  const int tid = threadIdx.x, lane = tid & 31, wave = tid >> 5;
  const int half = lane >> 4, l16 = lane & 15;
  const int qrow0 = blockIdx.x * 128 + wave * 16;

  __shared__ bf16_t Ks[32][DH + 8];      // [kv][dh], row 144B
  __shared__ bf16_t VsT[DH][32 + 8];     // transposed: [dh][kv], row 80B
  __shared__ bf16_t Ps[8][16][32 + 8];   // per-wave P transpose scratch

  // Q fragments in registers (two 32-wide k-steps over DH)
  frag16 qa[2];
  {
    const bf16_t* qp = Q + qbase + (size_t)(qrow0 + l16) * Dm;
    #pragma unroll
    for (int ks = 0; ks < 2; ++ks) {
      qa[ks].h[0] = *(const bf16x8*)(qp + ks * 32 + 8 * half);
      qa[ks].h[1] = *(const bf16x8*)(qp + ks * 32 + 16 + 8 * half);
    }
  }

  float mrow[8], lrow[8];
  f32x8 o[4] = {};
  #pragma unroll
  for (int r = 0; r < 8; ++r) { mrow[r] = -1e30f; lrow[r] = 0.f; }

  for (int kv0 = 0; kv0 < Skv; kv0 += 32) {
    __syncthreads();
    {                                     // K tile: 256 chunks, 1/thread (async)
      int r = tid >> 3, c8 = (tid & 7) << 3;
      async_cp16(K + kbase + (size_t)(kv0 + r) * Dm + c8, &Ks[r][c8]);
      // V tile transposed (sync; 8 scalar LDS stores)
      bf16x8 vv = *(const bf16x8*)(V + kbase + (size_t)(kv0 + r) * Dm + c8);
      #pragma unroll
      for (int jj = 0; jj < 8; ++jj) VsT[c8 + jj][r] = vv[jj];
    }
    async_wait0();
    __syncthreads();

    // scores S(16x32): 2 col-blocks x 2 k-steps
    f32x8 sc[2] = {};
    #pragma unroll
    for (int jb = 0; jb < 2; ++jb)
      #pragma unroll
      for (int ks = 0; ks < 2; ++ks) {
        frag16 kb;
        kb.h[0] = *(const bf16x8*)&Ks[jb * 16 + l16][ks * 32 + 8 * half];
        kb.h[1] = *(const bf16x8*)&Ks[jb * 16 + l16][ks * 32 + 16 + 8 * half];
        sc[jb] = __builtin_amdgcn_wmma_f32_16x16x32_bf16(
            false, qa[ks].f, false, kb.f, (short)0, sc[jb], false, false);
      }

    // online softmax; row r+8*half spans 16 lanes of this half
    float p0[8], p1[8];
    #pragma unroll
    for (int r = 0; r < 8; ++r) {
      float mx = redmax16(fmaxf(sc[0][r], sc[1][r]));
      const float mnew = fmaxf(mrow[r], mx);
      const float corr = __expf(mrow[r] - mnew);
      mrow[r] = mnew;
      const float e0 = __expf(sc[0][r] - mnew);
      const float e1 = __expf(sc[1][r] - mnew);
      p0[r] = e0; p1[r] = e1;
      lrow[r] = lrow[r] * corr + redsum16(e0 + e1);
      #pragma unroll
      for (int t = 0; t < 4; ++t) o[t][r] *= corr;
    }

    // transpose P (C layout -> A layout) via wave-private LDS (in-order DS)
    #pragma unroll
    for (int r = 0; r < 8; ++r) {
      Ps[wave][r + 8 * half][l16]      = f2bf(p0[r]);
      Ps[wave][r + 8 * half][16 + l16] = f2bf(p1[r]);
    }
    frag16 pa;
    pa.h[0] = *(const bf16x8*)&Ps[wave][l16][8 * half];
    pa.h[1] = *(const bf16x8*)&Ps[wave][l16][16 + 8 * half];

    // O(16x64) += P(16x32) . V(32x64), B-fragments contiguous in VsT
    #pragma unroll
    for (int t = 0; t < 4; ++t) {
      frag16 vbf;
      vbf.h[0] = *(const bf16x8*)&VsT[t * 16 + l16][8 * half];
      vbf.h[1] = *(const bf16x8*)&VsT[t * 16 + l16][16 + 8 * half];
      o[t] = __builtin_amdgcn_wmma_f32_16x16x32_bf16(
          false, pa.f, false, vbf.f, (short)0, o[t], false, false);
    }
  }

  float rcp[8];
  #pragma unroll
  for (int r = 0; r < 8; ++r) rcp[r] = 1.0f / lrow[r];
  #pragma unroll
  for (int t = 0; t < 4; ++t)
    #pragma unroll
    for (int r = 0; r < 8; ++r) {
      const int row = qrow0 + r + 8 * half;
      O[qbase + (size_t)row * Dm + t * 16 + l16] = f2bf(o[t][r] * rcp[r]);
    }
}

// ===========================================================================
// Host orchestration
// ===========================================================================
enum {
  IN_SRC = 0, IN_TGT,
  E_WQ, E_BQ, E_WK, E_BK, E_WV, E_BV, E_WO, E_BO,
  E_W1, E_B1, E_W2, E_B2, E_N1G, E_N1B, E_N2G, E_N2B,
  D_SAWQ, D_SABQ, D_SAWK, D_SABK, D_SAWV, D_SABV, D_SAWO, D_SABO,
  D_CAWQ, D_CABQ, D_CAWK, D_CABK, D_CAWV, D_CABV, D_CAWO, D_CABO,
  D_W1, D_B1, D_W2, D_B2, D_N1G, D_N1B, D_N2G, D_N2B, D_N3G, D_N3B,
  IN_ENG, IN_ENB, IN_DNG, IN_DNB
};

extern "C" void kernel_launch(void* const* d_in, const int* in_sizes, int n_in,
                              void* d_out, int out_size, void* d_ws, size_t ws_size,
                              hipStream_t stream) {
  (void)in_sizes; (void)n_in; (void)out_size; (void)ws_size;

  constexpr size_t MB = 1024 * 1024;
  char* wsb = (char*)d_ws;
  float*  x    = (float*)(wsb + 0 * MB);    // encoder state (fp32)
  float*  y    = (float*)(wsb + 16 * MB);   // decoder state (fp32)
  bf16_t* hb   = (bf16_t*)(wsb + 32 * MB);  // LN output (bf16)
  bf16_t* qb   = (bf16_t*)(wsb + 40 * MB);
  bf16_t* kbuf = (bf16_t*)(wsb + 48 * MB);
  bf16_t* vbuf = (bf16_t*)(wsb + 56 * MB);
  bf16_t* aob  = (bf16_t*)(wsb + 64 * MB);  // attention out (bf16)
  bf16_t* ffb  = (bf16_t*)(wsb + 72 * MB);  // FFN mid (bf16)
  bf16_t* memb = (bf16_t*)(wsb + 80 * MB);  // encoder memory (bf16)
  bf16_t* wcb  = (bf16_t*)(wsb + 88 * MB);  // converted weight (2MB)

  auto P  = [&](int idx) { return (const float*)d_in[idx]; };
  auto PW = [&](int idx, int l) { return (const float*)d_in[idx] + (size_t)l * Dm * Dm; };
  auto PB = [&](int idx, int l) { return (const float*)d_in[idx] + (size_t)l * Dm; };

  const dim3 gGemm(Dm / BN, Mr / BM), gAttn(Ss / 128, Bb * Hh), blk(256);

  auto CVTW = [&](const float* w) {
    cvt_kernel<<<(Dm * Dm) / 1024, blk, 0, stream>>>(w, wcb, Dm * Dm);
  };
  auto LNB = [&](const float* in, const float* g, const float* b, bf16_t* out) {
    ln_kernel<bf16_t><<<Mr, blk, 0, stream>>>(in, g, b, out);
  };
  auto LNF = [&](const float* in, const float* g, const float* b, float* out) {
    ln_kernel<float><<<Mr, blk, 0, stream>>>(in, g, b, out);
  };
  auto GEMM = [&](const bf16_t* X, const float* bias, bf16_t* out, float alpha) {
    gemm_kernel<0, false, bf16_t><<<gGemm, blk, 0, stream>>>(
        X, wcb, bias, nullptr, out, Mr, Dm, Dm, alpha);
  };
  auto GEMM_RELU = [&](const bf16_t* X, const float* bias, bf16_t* out) {
    gemm_kernel<1, false, bf16_t><<<gGemm, blk, 0, stream>>>(
        X, wcb, bias, nullptr, out, Mr, Dm, Dm, 1.0f);
  };
  auto GEMM_RES = [&](const bf16_t* X, const float* bias, const float* res, float* out) {
    gemm_kernel<0, true, float><<<gGemm, blk, 0, stream>>>(
        X, wcb, bias, res, out, Mr, Dm, Dm, 1.0f);
  };
  auto ATTN = [&]() {
    flash_attn_kernel<<<gAttn, blk, 0, stream>>>(qb, kbuf, vbuf, aob, Ss, Ss);
  };

  (void)hipMemcpyAsync(x, d_in[IN_SRC], (size_t)Mr * Dm * 4, hipMemcpyDeviceToDevice, stream);
  (void)hipMemcpyAsync(y, d_in[IN_TGT], (size_t)Mr * Dm * 4, hipMemcpyDeviceToDevice, stream);

  // ----------------- encoder -----------------
  for (int i = 0; i < 6; ++i) {
    LNB(x, PB(E_N1G, i), PB(E_N1B, i), hb);
    CVTW(PW(E_WQ, i)); GEMM(hb, PB(E_BQ, i), qb,   0.125f);   // alpha = 1/sqrt(DH)
    CVTW(PW(E_WK, i)); GEMM(hb, PB(E_BK, i), kbuf, 1.0f);
    CVTW(PW(E_WV, i)); GEMM(hb, PB(E_BV, i), vbuf, 1.0f);
    ATTN();
    CVTW(PW(E_WO, i)); GEMM_RES(aob, PB(E_BO, i), x, x);
    LNB(x, PB(E_N2G, i), PB(E_N2B, i), hb);
    CVTW(PW(E_W1, i)); GEMM_RELU(hb, PB(E_B1, i), ffb);
    CVTW(PW(E_W2, i)); GEMM_RES(ffb, PB(E_B2, i), x, x);
  }
  LNB(x, P(IN_ENG), P(IN_ENB), memb);

  // ----------------- decoder -----------------
  for (int i = 0; i < 6; ++i) {
    LNB(y, PB(D_N1G, i), PB(D_N1B, i), hb);
    CVTW(PW(D_SAWQ, i)); GEMM(hb, PB(D_SABQ, i), qb,   0.125f);
    CVTW(PW(D_SAWK, i)); GEMM(hb, PB(D_SABK, i), kbuf, 1.0f);
    CVTW(PW(D_SAWV, i)); GEMM(hb, PB(D_SABV, i), vbuf, 1.0f);
    ATTN();
    CVTW(PW(D_SAWO, i)); GEMM_RES(aob, PB(D_SABO, i), y, y);

    LNB(y, PB(D_N2G, i), PB(D_N2B, i), hb);
    CVTW(PW(D_CAWQ, i)); GEMM(hb,   PB(D_CABQ, i), qb,   0.125f);
    CVTW(PW(D_CAWK, i)); GEMM(memb, PB(D_CABK, i), kbuf, 1.0f);
    CVTW(PW(D_CAWV, i)); GEMM(memb, PB(D_CABV, i), vbuf, 1.0f);
    ATTN();
    CVTW(PW(D_CAWO, i)); GEMM_RES(aob, PB(D_CABO, i), y, y);

    LNB(y, PB(D_N3G, i), PB(D_N3B, i), hb);
    CVTW(PW(D_W1, i)); GEMM_RELU(hb, PB(D_B1, i), ffb);
    CVTW(PW(D_W2, i)); GEMM_RES(ffb, PB(D_B2, i), y, y);
  }
  LNF(y, P(IN_DNG), P(IN_DNB), (float*)d_out);
}
